// TextGuidedCamFusion_76794015252558
// MI455X (gfx1250) — matrix operand order repaired and checked by reading, not compiled, not run on hardware
//
#include <hip/hip_runtime.h>
#include <hip/hip_bf16.h>
#include <math.h>

// ---------------------------------------------------------------------------
// MI455X (gfx1250, wave32) fused pipeline for TextGuidedCamFusion.
// Dominant GEMM runs on v_wmma_f32_16x16x32_bf16 (f32 accumulate).
// ---------------------------------------------------------------------------

typedef __attribute__((ext_vector_type(16))) __bf16 bf16x16;
typedef __attribute__((ext_vector_type(8)))  float  f32x8;

#define BLOCK    256
#define WAVES    8
#define MROWS    32        // rows (batch) per workgroup
#define DOUT     512       // D
#define KTEXT    8         // K text vectors
#define NLEVEL   4
#define ASTRIDE  80        // bytes per A row in LDS: 32 bf16 (64B) + 16B pad (bank-conflict-free b128)
#define BSTRIDE  80        // bytes per B column in LDS: 16 packed k-pair dwords (64B) + 16B pad

// ---------------------------------------------------------------------------
// Prep: L2-normalize text_base [8,512] -> ws
// ---------------------------------------------------------------------------
__global__ __launch_bounds__(256)
void text_norm_kernel(const float* __restrict__ tb, float* __restrict__ tf) {
  __shared__ float part[WAVES];
  const int k = blockIdx.x;          // 0..7
  const int t = threadIdx.x;
  float x0 = tb[k * DOUT + t];
  float x1 = tb[k * DOUT + 256 + t];
  float q = x0 * x0 + x1 * x1;
  q += __shfl_xor(q, 1, 32);  q += __shfl_xor(q, 2, 32);
  q += __shfl_xor(q, 4, 32);  q += __shfl_xor(q, 8, 32);
  q += __shfl_xor(q, 16, 32);
  if ((t & 31) == 0) part[t >> 5] = q;
  __syncthreads();
  float S = 0.f;
  for (int i = 0; i < WAVES; ++i) S += part[i];
  float inv = 1.f / fmaxf(sqrtf(S), 1e-12f);
  tf[k * DOUT + t]       = x0 * inv;
  tf[k * DOUT + 256 + t] = x1 * inv;
}

// ---------------------------------------------------------------------------
// Prep: proto_mean[c] = mean over P=64 prototype rows
// ---------------------------------------------------------------------------
__global__ __launch_bounds__(256)
void proto_mean_kernel(const float* __restrict__ proto, float* __restrict__ pm, int C) {
  const int c = blockIdx.x * 256 + threadIdx.x;
  if (c < C) {
    float s = 0.f;
    for (int p = 0; p < 64; ++p) s += proto[(size_t)p * C + c];
    pm[c] = s * (1.f / 64.f);
  }
}

// Prep: scalar sum / sumsq of proto_mean (deterministic single-block reduce)
__global__ __launch_bounds__(256)
void proto_stats_kernel(const float* __restrict__ pm, float* __restrict__ st, int C) {
  __shared__ float ps[WAVES], qs[WAVES];
  const int t = threadIdx.x;
  float s = 0.f, q = 0.f;
  for (int c = t; c < C; c += 256) { float x = pm[c]; s += x; q += x * x; }
  s += __shfl_xor(s, 1, 32);  q += __shfl_xor(q, 1, 32);
  s += __shfl_xor(s, 2, 32);  q += __shfl_xor(q, 2, 32);
  s += __shfl_xor(s, 4, 32);  q += __shfl_xor(q, 4, 32);
  s += __shfl_xor(s, 8, 32);  q += __shfl_xor(q, 8, 32);
  s += __shfl_xor(s, 16, 32); q += __shfl_xor(q, 16, 32);
  if ((t & 31) == 0) { ps[t >> 5] = s; qs[t >> 5] = q; }
  __syncthreads();
  if (t == 0) {
    float S = 0.f, Q = 0.f;
    for (int i = 0; i < WAVES; ++i) { S += ps[i]; Q += qs[i]; }
    st[0] = S; st[1] = Q;
  }
}

// ---------------------------------------------------------------------------
// Main: LayerNorm(concat) -> bf16 WMMA GEMM (M=32,N=512,K=2C) -> bias/ReLU ->
//       L2-norm + text dot -> raw logits into d_out[b,k,level]
// One workgroup (8 waves) owns 32 rows x all 512 output cols.
// ---------------------------------------------------------------------------
__global__ __launch_bounds__(256)
void fuse_ln_gemm_logits(const float* __restrict__ pooled,      // [B, C]
                         const float* __restrict__ ln_g,        // [2C]
                         const float* __restrict__ ln_b,        // [2C]
                         const float* __restrict__ W,           // [2C, 512]
                         const float* __restrict__ bias,        // [512]
                         const float* __restrict__ proto_mean,  // [C]
                         const float* __restrict__ proto_stats, // {sum, sumsq}
                         const float* __restrict__ text,        // [8,512] normalized
                         const float* __restrict__ logit_scale, // [1]
                         float* __restrict__ out,               // [B,8,4]
                         int C, int level)
{
  __shared__ __align__(16) char ldsA[MROWS * ASTRIDE];   //  2.5 KB
  __shared__ __align__(16) char ldsB[DOUT * BSTRIDE];    // 40   KB
  __shared__ float smean[MROWS], srstd[MROWS];
  __shared__ float sdotw[WAVES][MROWS][KTEXT];           //  8   KB
  __shared__ float srowsq[WAVES][MROWS];                 //  1   KB

  const int t    = threadIdx.x;
  const int wv   = t >> 5;        // wave id 0..7 -> owns 64 output cols
  const int lane = t & 31;
  const int l16  = lane & 15;
  const int g    = lane >> 4;     // half-wave group (K split of WMMA fragments)
  const int row0 = blockIdx.x * MROWS;
  const int twoC = 2 * C;

  // ---- stage 0: LayerNorm stats per fused row (pooled part + precomputed proto part)
  {
    const int r = t >> 3;   // 8 threads per row
    const int j = t & 7;
    const float* prow = pooled + (size_t)(row0 + r) * C;
    float s = 0.f, q = 0.f;
    for (int c = j; c < C; c += 8) { float x = prow[c]; s += x; q += x * x; }
    s += __shfl_xor(s, 1, 32);  q += __shfl_xor(q, 1, 32);
    s += __shfl_xor(s, 2, 32);  q += __shfl_xor(q, 2, 32);
    s += __shfl_xor(s, 4, 32);  q += __shfl_xor(q, 4, 32);
    if (j == 0) {
      float m = (s + proto_stats[0]) / (float)twoC;
      float v = (q + proto_stats[1]) / (float)twoC - m * m;
      smean[r] = m;
      srstd[r] = rsqrtf(v + 1e-5f);
    }
  }
  __syncthreads();

  f32x8 acc[2][4];
  const f32x8 fzero = {0.f, 0.f, 0.f, 0.f, 0.f, 0.f, 0.f, 0.f};
  for (int rt = 0; rt < 2; ++rt)
    for (int ct = 0; ct < 4; ++ct) acc[rt][ct] = fzero;

  // ---- K loop: 32 fused-K per step, A/B converted to bf16 in LDS
  const int nk = twoC / 32;
  for (int kc = 0; kc < nk; ++kc) {
    const int k0 = kc * 32;

    // A tile: 32 rows x 16 bf16-pairs (normalize + affine + cvt)
    for (int i = 0; i < 2; ++i) {
      int pidx = t + i * BLOCK;            // 0..511
      int r = pidx >> 4, kp = pidx & 15;
      int kg = k0 + kp * 2;
      float m = smean[r], rs = srstd[r];
      float x0, x1;
      if (kg < C) {
        const float* p = pooled + (size_t)(row0 + r) * C + kg;
        x0 = p[0]; x1 = p[1];
      } else {
        const float* p = proto_mean + (kg - C);
        x0 = p[0]; x1 = p[1];
      }
      x0 = (x0 - m) * rs * ln_g[kg]     + ln_b[kg];
      x1 = (x1 - m) * rs * ln_g[kg + 1] + ln_b[kg + 1];
      union { __bf16 h[2]; unsigned u; } pk;
      pk.h[0] = (__bf16)x0; pk.h[1] = (__bf16)x1;
      *(unsigned*)(ldsA + r * ASTRIDE + kp * 4) = pk.u;
    }

    // B tile: per column n, 16 packed k-pair dwords (f32 W -> bf16)
    for (int i = 0; i < 32; ++i) {
      int idx = t + i * BLOCK;             // 0..8191
      int j = idx >> 9, n = idx & 511;     // j = k-pair 0..15
      const float* wp = W + (size_t)(k0 + 2 * j) * DOUT + n;
      union { __bf16 h[2]; unsigned u; } pk;
      pk.h[0] = (__bf16)wp[0];
      pk.h[1] = (__bf16)wp[DOUT];
      *(unsigned*)(ldsB + n * BSTRIDE + j * 4) = pk.u;
    }
    __syncthreads();

    // fragments per ISA 16-bit layouts, then 2x4 WMMA
    union FragA { uint4 q[2]; bf16x16 v; } fa[2];
    union FragB { uint4 q[2]; bf16x16 v; } fb[4];
    for (int rt = 0; rt < 2; ++rt) {
      const char* ap = ldsA + (rt * 16 + l16) * ASTRIDE + g * 16;
      fa[rt].q[0] = *(const uint4*)(ap);
      fa[rt].q[1] = *(const uint4*)(ap + 32);
    }
    for (int ct = 0; ct < 4; ++ct) {
      int n = (wv * 4 + ct) * 16 + l16;
      const char* bp = ldsB + n * BSTRIDE + g * 16;
      fb[ct].q[0] = *(const uint4*)(bp);
      fb[ct].q[1] = *(const uint4*)(bp + 32);
    }
    for (int rt = 0; rt < 2; ++rt)
      for (int ct = 0; ct < 4; ++ct)
        acc[rt][ct] = __builtin_amdgcn_wmma_f32_16x16x32_bf16(
            false, fa[rt].v, false, fb[ct].v, (short)0, acc[rt][ct], false, false);
    __syncthreads();
  }

  // ---- epilogue: bias + ReLU (in registers)
  {
    float bcol[4];
    for (int ct = 0; ct < 4; ++ct) bcol[ct] = bias[(wv * 4 + ct) * 16 + l16];
    for (int rt = 0; rt < 2; ++rt)
      for (int ct = 0; ct < 4; ++ct)
        for (int a = 0; a < 8; ++a)
          acc[rt][ct][a] = fmaxf(acc[rt][ct][a] + bcol[ct], 0.f);
  }

  // per-row ||h||^2: 16 independent chains batched per xor-step so the
  // ds_bpermute waves pipeline instead of serializing on s_wait_dscnt 0.
  {
    float sq16[16];
    for (int rt = 0; rt < 2; ++rt)
      for (int a = 0; a < 8; ++a) {
        float s = 0.f;
        for (int ct = 0; ct < 4; ++ct) { float h = acc[rt][ct][a]; s += h * h; }
        sq16[rt * 8 + a] = s;
      }
    for (int step = 1; step <= 8; step <<= 1)
      for (int i = 0; i < 16; ++i)
        sq16[i] += __shfl_xor(sq16[i], step, 32);
    if (l16 == 0)
      for (int rt = 0; rt < 2; ++rt)
        for (int a = 0; a < 8; ++a)
          srowsq[wv][rt * 16 + a + g * 8] = sq16[rt * 8 + a];
  }

  // text dots: k outer (only 4 text values live vs. the 64 acc VGPRs -> no
  // spill); 16 row-chains batched per xor-step for pipelined ds_bpermute.
  for (int k = 0; k < KTEXT; ++k) {
    float tvk[4];
    for (int ct = 0; ct < 4; ++ct) tvk[ct] = text[k * DOUT + (wv * 4 + ct) * 16 + l16];
    float dp16[16];
    for (int rt = 0; rt < 2; ++rt)
      for (int a = 0; a < 8; ++a) {
        float d = 0.f;
        for (int ct = 0; ct < 4; ++ct) d += acc[rt][ct][a] * tvk[ct];
        dp16[rt * 8 + a] = d;
      }
    for (int step = 1; step <= 8; step <<= 1)
      for (int i = 0; i < 16; ++i)
        dp16[i] += __shfl_xor(dp16[i], step, 32);
    if (l16 == 0)
      for (int rt = 0; rt < 2; ++rt)
        for (int a = 0; a < 8; ++a)
          sdotw[wv][rt * 16 + a + g * 8][k] = dp16[rt * 8 + a];
  }
  __syncthreads();

  // ---- combine across waves and write raw (pre-softmax) logits
  {
    const int row = t >> 3, k = t & 7;
    float nrm2 = 0.f, dp = 0.f;
    for (int w = 0; w < WAVES; ++w) { nrm2 += srowsq[w][row]; dp += sdotw[w][row][k]; }
    float denom = fmaxf(sqrtf(nrm2), 1e-12f);              // L2-norm folded into dot
    float ls = fmaxf(logit_scale[0], 1e-4f);
    float val = dp / denom * ls * 0.04419417382415922f;    // 1/sqrt(512)
    out[((size_t)(row0 + row) * KTEXT + k) * NLEVEL + level] = val;
  }
}

// ---------------------------------------------------------------------------
// Final: softmax over the 4 levels, in place on d_out
// ---------------------------------------------------------------------------
__global__ __launch_bounds__(256)
void softmax_l_kernel(float* __restrict__ out, int n) {
  int i = blockIdx.x * blockDim.x + threadIdx.x;
  if (i < n) {
    float4 v = *(float4*)(out + (size_t)i * 4);
    float m = fmaxf(fmaxf(v.x, v.y), fmaxf(v.z, v.w));
    float e0 = __expf(v.x - m), e1 = __expf(v.y - m);
    float e2 = __expf(v.z - m), e3 = __expf(v.w - m);
    float inv = 1.f / (e0 + e1 + e2 + e3);
    float4 r = {e0 * inv, e1 * inv, e2 * inv, e3 * inv};
    *(float4*)(out + (size_t)i * 4) = r;
  }
}

// ---------------------------------------------------------------------------
// Launch
// Input order (setup_inputs dict order): per level i: [6i]=pooled, [6i+1]=proto,
// [6i+2]=ln_g, [6i+3]=ln_b, [6i+4]=W, [6i+5]=b; [24]=text_base, [25]=logit_scale.
// ---------------------------------------------------------------------------
extern "C" void kernel_launch(void* const* d_in, const int* in_sizes, int n_in,
                              void* d_out, int out_size, void* d_ws, size_t ws_size,
                              hipStream_t stream) {
  (void)in_sizes; (void)n_in; (void)out_size; (void)ws_size;
  static const int LEVEL_DIMS[NLEVEL] = {256, 512, 1024, 2048};
  const int B = 8192;

  float* ws          = (float*)d_ws;
  float* text_feat   = ws;                     // 4096 floats
  float* proto_mean  = ws + 4096;              // 3840 floats total
  float* proto_stats = ws + 4096 + 3840;       // 8 floats (sum,sumsq per level)

  text_norm_kernel<<<KTEXT, 256, 0, stream>>>((const float*)d_in[24], text_feat);

  int pmoff = 0;
  for (int l = 0; l < NLEVEL; ++l) {
    int C = LEVEL_DIMS[l];
    proto_mean_kernel<<<(C + 255) / 256, 256, 0, stream>>>(
        (const float*)d_in[6 * l + 1], proto_mean + pmoff, C);
    proto_stats_kernel<<<1, 256, 0, stream>>>(proto_mean + pmoff, proto_stats + 2 * l, C);
    pmoff += C;
  }

  pmoff = 0;
  for (int l = 0; l < NLEVEL; ++l) {
    int C = LEVEL_DIMS[l];
    fuse_ln_gemm_logits<<<B / MROWS, BLOCK, 0, stream>>>(
        (const float*)d_in[6 * l + 0],   // pooled
        (const float*)d_in[6 * l + 2],   // ln_g
        (const float*)d_in[6 * l + 3],   // ln_b
        (const float*)d_in[6 * l + 4],   // W
        (const float*)d_in[6 * l + 5],   // bias
        proto_mean + pmoff, proto_stats + 2 * l,
        text_feat, (const float*)d_in[25],
        (float*)d_out, C, l);
    pmoff += C;
  }

  softmax_l_kernel<<<(B * KTEXT + 255) / 256, 256, 0, stream>>>((float*)d_out, B * KTEXT);
}
